// _S2M2_79379585565399
// MI455X (gfx1250) — matrix operand order repaired and checked by reading, not compiled, not run on hardware
//
#include <hip/hip_runtime.h>

typedef __attribute__((ext_vector_type(16))) _Float16 v16h;
typedef __attribute__((ext_vector_type(8)))  _Float16 v8h;
typedef __attribute__((ext_vector_type(8)))  float    v8f;

namespace {
constexpr int B2  = 4;      // 2*B
constexpr int Bb  = 2;      // B
constexpr int C   = 128;
constexpr int H   = 184;
constexpr int W   = 320;
constexpr int HW  = H * W;          // 58880
constexpr int BHW = Bb * H * W;     // 117760
constexpr float LOG2W  = 6.4614681f;    //  ln(640)
constexpr float LOGM   = -6.4614681f;   // -ln(640)  (marginal 1/(2w))
constexpr float LOGML  = -0.6931472f;   //  ln(0.5)  (last marginal w/(2w))
}

// ---------------------------------------------------------------------------
// Kernel 1: channel LayerNorm + fp32 -> f16, [2B,C,H,W] -> [2B,H,W,C]
// ---------------------------------------------------------------------------
__global__ __launch_bounds__(256) void ln_kernel(const float* __restrict__ feat,
                                                 const float* __restrict__ gamma,
                                                 const float* __restrict__ beta,
                                                 _Float16* __restrict__ out) {
  __shared__ float tile[64][C + 1];   // +1 pad: avoid 64-way bank conflicts
  const int blk = blockIdx.x;         // B2*H*(W/64) blocks
  const int w0  = (blk % (W / 64)) * 64;
  const int bh  = blk / (W / 64);     // b2*H + h
  const int t   = threadIdx.x;

  const float* base = feat + (size_t)(bh / H) * C * HW + (size_t)(bh % H) * W + w0;

  // coalesced load: 64 pixels x 128 channels
  #pragma unroll
  for (int it = 0; it < 32; ++it) {
    int idx = it * 256 + t;
    int c = idx >> 6, wl = idx & 63;
    tile[wl][c] = base[(size_t)c * HW + wl];
  }
  __syncthreads();

  const int lane = t & 31, wv = t >> 5;
  #pragma unroll
  for (int pp = 0; pp < 8; ++pp) {
    const int p = wv * 8 + pp;
    float x0 = tile[p][lane +  0];
    float x1 = tile[p][lane + 32];
    float x2 = tile[p][lane + 64];
    float x3 = tile[p][lane + 96];
    float s  = x0 + x1 + x2 + x3;
    float s2 = x0 * x0 + x1 * x1 + x2 * x2 + x3 * x3;
    #pragma unroll
    for (int off = 16; off; off >>= 1) {
      s  += __shfl_xor(s,  off, 32);
      s2 += __shfl_xor(s2, off, 32);
    }
    const float mu  = s * (1.0f / C);
    const float var = s2 * (1.0f / C) - mu * mu;
    const float rs  = rsqrtf(var + 1e-5f);
    _Float16* orow = out + ((size_t)bh * W + w0 + p) * C;
    orow[lane +  0] = (_Float16)((x0 - mu) * rs * gamma[lane +  0] + beta[lane +  0]);
    orow[lane + 32] = (_Float16)((x1 - mu) * rs * gamma[lane + 32] + beta[lane + 32]);
    orow[lane + 64] = (_Float16)((x2 - mu) * rs * gamma[lane + 64] + beta[lane + 64]);
    orow[lane + 96] = (_Float16)((x3 - mu) * rs * gamma[lane + 96] + beta[lane + 96]);
  }
}

// ---------------------------------------------------------------------------
// Kernel 2: per-(b,h) cost volume via WMMA: cv = f0row (320x128) x f1row^T
// One workgroup (8 waves) per (b,h). A,B staged in 160KB LDS via async
// global->LDS copies (ASYNCcnt path); each wave computes 32x32 macro-tiles
// (2x2 WMMA register blocking, 16 WMMAs per macro-tile).
// ---------------------------------------------------------------------------
__device__ __forceinline__ v16h load_a_frag(const _Float16* ap) {
  // A 16x32 f16 frag: lane(m) holds K = hsel*16 + 0..15;
  // ap already points at row + kk*32 + hsel*16
  v8h lo = *(const v8h*)ap;
  v8h hi = *(const v8h*)(ap + 8);
  v16h a;
  #pragma unroll
  for (int q = 0; q < 8; ++q) { a[q] = lo[q]; a[q + 8] = hi[q]; }
  return a;
}

__global__ __launch_bounds__(256) void cv_kernel(const _Float16* __restrict__ f,
                                                 float* __restrict__ cvout) {
  extern __shared__ _Float16 sm[];
  _Float16* As = sm;           // [320][128]  rows of f0
  _Float16* Bs = sm + W * C;   // [320][128]  rows of f1

  const int bh = blockIdx.x;   // b*H + h
  const uint4* ga = (const uint4*)(f + (size_t)bh * W * C);
  const uint4* gb = (const uint4*)(f + (size_t)(bh + Bb * H) * W * C);

  // Stage A and B via async global->LDS (no VGPR round-trip, ASYNCcnt).
  // Dynamic LDS starts at group offset 0: raw byte offsets are LDS addresses.
  for (int i = threadIdx.x; i < W * C / 8; i += 256) {
    unsigned int la = (unsigned int)(i * 16);
    unsigned int lb = (unsigned int)(W * C * 2 + i * 16);
    unsigned long long pa = (unsigned long long)(ga + i);
    unsigned long long pb = (unsigned long long)(gb + i);
    asm volatile("global_load_async_to_lds_b128 %0, %1, off"
                 :: "v"(la), "v"(pa) : "memory");
    asm volatile("global_load_async_to_lds_b128 %0, %1, off"
                 :: "v"(lb), "v"(pb) : "memory");
  }
  asm volatile("s_wait_asynccnt 0x0" ::: "memory");
  __syncthreads();

  const int lane = threadIdx.x & 31;
  const int wv   = threadIdx.x >> 5;
  const int l16  = lane & 15;   // M%16 for A, N%16 for B, N for C/D
  const int hsel = lane >> 4;   // K half-select
  float* outb = cvout + (size_t)bh * W * W;

  constexpr int NT = W / 32;    // 10 macro-tiles per dimension
  for (int mt = wv; mt < NT * NT; mt += 8) {
    const int mi = (mt / NT) * 32;
    const int nj = (mt % NT) * 32;
    v8f acc00 = {0.f,0.f,0.f,0.f,0.f,0.f,0.f,0.f};
    v8f acc01 = {0.f,0.f,0.f,0.f,0.f,0.f,0.f,0.f};
    v8f acc10 = {0.f,0.f,0.f,0.f,0.f,0.f,0.f,0.f};
    v8f acc11 = {0.f,0.f,0.f,0.f,0.f,0.f,0.f,0.f};
    const _Float16* a0row = As + (size_t)(mi +      l16) * C + hsel * 16;
    const _Float16* a1row = As + (size_t)(mi + 16 + l16) * C + hsel * 16;
    const _Float16* b0row = Bs + (size_t)(nj +      l16) * C + hsel * 16;
    const _Float16* b1row = Bs + (size_t)(nj + 16 + l16) * C + hsel * 16;
    #pragma unroll
    for (int kk = 0; kk < C / 32; ++kk) {
      v16h a0 = load_a_frag(a0row + kk * 32);
      v16h a1 = load_a_frag(a1row + kk * 32);
      // B 32x16 f16 frag: lane(n) holds K = hsel*16 + 0..15 (row-major f1 row)
      v16h b0 = *(const v16h*)(b0row + kk * 32);
      v16h b1 = *(const v16h*)(b1row + kk * 32);
      acc00 = __builtin_amdgcn_wmma_f32_16x16x32_f16(false, a0, false, b0,
                                                     (short)0, acc00, false, false);
      acc01 = __builtin_amdgcn_wmma_f32_16x16x32_f16(false, a0, false, b1,
                                                     (short)0, acc01, false, false);
      acc10 = __builtin_amdgcn_wmma_f32_16x16x32_f16(false, a1, false, b0,
                                                     (short)0, acc10, false, false);
      acc11 = __builtin_amdgcn_wmma_f32_16x16x32_f16(false, a1, false, b1,
                                                     (short)0, acc11, false, false);
    }
    // C/D 16x16 f32: VGPR r -> M = r + 8*hsel, N = l16
    #pragma unroll
    for (int r = 0; r < 8; ++r) {
      const int row0 = mi + r + hsel * 8;
      outb[(size_t)row0 * W        + nj      + l16] = acc00[r];
      outb[(size_t)row0 * W        + nj + 16 + l16] = acc01[r];
      outb[(size_t)(row0 + 16) * W + nj      + l16] = acc10[r];
      outb[(size_t)(row0 + 16) * W + nj + 16 + l16] = acc11[r];
    }
  }
}

// ---------------------------------------------------------------------------
// Kernel 3: Sinkhorn (5 iters, log-space) + soft-argmax refinement.
// One workgroup per (b,h); cv read from L2 (151MB < 192MB L2); u,v in LDS.
// ---------------------------------------------------------------------------
__device__ __forceinline__ void lse_push(float& m, float& s, float x) {
  float mn = fmaxf(m, x);
  s = s * __expf(m - mn) + __expf(x - mn);
  m = mn;
}
__device__ __forceinline__ void lse_wave_reduce(float& m, float& s) {
  #pragma unroll
  for (int off = 16; off; off >>= 1) {
    float mo = __shfl_xor(m, off, 32);
    float so = __shfl_xor(s, off, 32);
    float mn = fmaxf(m, mo);
    s = s * __expf(m - mn) + so * __expf(mo - mn);
    m = mn;
  }
}

__global__ __launch_bounds__(256) void sinkhorn_kernel(const float* __restrict__ cv,
                                                       float* __restrict__ disp,
                                                       float* __restrict__ conf_o,
                                                       float* __restrict__ occ_o) {
  __shared__ float u[W + 1];
  __shared__ float v[W + 1];
  const int bh = blockIdx.x;
  const float* crow = cv + (size_t)bh * W * W;
  const int t = threadIdx.x, lane = t & 31, wv = t >> 5;

  for (int j = t; j <= W; j += 256) u[j] = 0.0f;
  __syncthreads();

  for (int it = 0; it < 5; ++it) {
    // ---- v[j] = log_m[j] - lse_i(attn[i][j] + u[i]); thread-per-column ----
    for (int j = t; j <= W; j += 256) {
      float m = -1e30f, s = 0.0f;
      if (j < W) {
        for (int i = 0; i < W; ++i) {            // uniform i: coalesced row reads
          if (i >= j)                             // j>i is NEG -> exp underflows to 0
            lse_push(m, s, crow[(size_t)i * W + j] + u[i]);
        }
        lse_push(m, s, u[W]);                     // pad row: attn==0
        v[j] = LOGM - (m + logf(fmaxf(s, 1e-30f)));
      } else {                                    // pad column j==W: attn==0
        for (int i = 0; i <= W; ++i) lse_push(m, s, u[i]);
        v[W] = LOGML - (m + logf(fmaxf(s, 1e-30f)));
      }
    }
    __syncthreads();
    // ---- u[i] = log_m[i] - lse_j(attn[i][j] + v[j]); wave-per-row ----
    for (int i = wv; i < W; i += 8) {
      float m = -1e30f, s = 0.0f;
      for (int j = lane; j <= i; j += 32)
        lse_push(m, s, crow[(size_t)i * W + j] + v[j]);
      if (lane == 0) lse_push(m, s, v[W]);        // pad col
      lse_wave_reduce(m, s);
      if (lane == 0) u[i] = LOGM - (m + logf(fmaxf(s, 1e-30f)));
    }
    if (wv == 0) {                                // pad row i==W
      float m = -1e30f, s = 0.0f;
      for (int j = lane; j <= W; j += 32) lse_push(m, s, v[j]);
      lse_wave_reduce(m, s);
      if (lane == 0) u[W] = LOGML - (m + logf(fmaxf(s, 1e-30f)));
    }
    __syncthreads();
  }

  // ---- final: occ, argmax window -> conf/corr/disparity; wave-per-row ----
  for (int i = wv; i < W; i += 8) {
    const float ui = u[i];
    float pmax = -1.0f, sum = 0.0f;
    int pidx = 0;
    for (int j = lane; j < W; j += 32) {
      float val = 0.0f;
      if (j <= i) val = __expf(crow[(size_t)i * W + j] + ui + v[j] + LOG2W);
      sum += val;
      if (val > pmax) { pmax = val; pidx = j; }
    }
    #pragma unroll
    for (int off = 16; off; off >>= 1) {
      sum += __shfl_xor(sum, off, 32);
      float om = __shfl_xor(pmax, off, 32);
      int   oi = __shfl_xor(pidx, off, 32);
      if (om > pmax || (om == pmax && oi < pidx)) { pmax = om; pidx = oi; }
    }
    if (lane == 0) {
      float cf = 0.0f, cr = 0.0f;
      #pragma unroll
      for (int d = -2; d <= 2; ++d) {
        int jj = pidx + d;
        float wgt = 0.0f;
        if (jj >= 0 && jj < W && jj <= i)
          wgt = __expf(crow[(size_t)i * W + jj] + ui + v[jj] + LOG2W);
        cf += wgt;
        cr += wgt * (float)jj;
      }
      cr = (cr + 1e-4f) / (cf + 1e-4f);
      const size_t o = (size_t)bh * W + i;
      disp[o]   = (float)i - cr;
      conf_o[o] = cf;
      occ_o[o]  = sum;
    }
  }
}

// ---------------------------------------------------------------------------
extern "C" void kernel_launch(void* const* d_in, const int* in_sizes, int n_in,
                              void* d_out, int out_size, void* d_ws, size_t ws_size,
                              hipStream_t stream) {
  const float* feat  = (const float*)d_in[0];
  const float* gamma = (const float*)d_in[1];
  const float* beta  = (const float*)d_in[2];

  float* out  = (float*)d_out;
  float* disp = out;                       // [B,1,H,W]
  float* conf = out + (size_t)BHW;         // [B,1,H,W]
  float* occ  = out + (size_t)2 * BHW;     // [B,1,H,W]
  float* cv   = out + (size_t)3 * BHW;     // [B,H,W,W]

  _Float16* fn = (_Float16*)d_ws;          // normalized f16 features [2B,H,W,C]

  ln_kernel<<<B2 * H * (W / 64), 256, 0, stream>>>(feat, gamma, beta, fn);
  cv_kernel<<<Bb * H, 256, 2 * W * C * sizeof(_Float16), stream>>>(fn, cv);
  sinkhorn_kernel<<<Bb * H, 256, 0, stream>>>(cv, disp, conf, occ);
}
